// FIVO_75831942578665
// MI455X (gfx1250) — matrix operand order repaired and checked by reading, not compiled, not run on hardware
//
#include <hip/hip_runtime.h>
#include <hip/hip_bf16.h>

#define DEVI static __device__ __forceinline__

typedef __attribute__((ext_vector_type(16))) __bf16 v16bf;
typedef __attribute__((ext_vector_type(8)))  float  v8f;

namespace {
constexpr int Bk = 32, Tk = 128, Dk = 2, Ek = 64, Lk = 64, Hk = 256, Pk = 128;
constexpr int INk = Ek + Lk + Dk;       // 130
constexpr int KIN = 160;                // K padded to multiple of 32
constexpr float SPB    = 0.5413248538970947f;
constexpr float LOG2PI = 1.8378770664093453f;
constexpr float LOGP   = 4.8520302639196169f;  // log(128)

// ---- workspace layout (bytes) ----
constexpr size_t OFF_WIH  = 0;
constexpr size_t SZ_WIH   = size_t(3*Hk)*KIN*2;   // bf16 [768][160]
constexpr size_t OFF_WHH  = OFF_WIH + SZ_WIH;
constexpr size_t SZ_WHH   = size_t(3*Hk)*Hk*2;    // bf16 [768][256]
constexpr size_t OFF_WENC = OFF_WHH + SZ_WHH;
constexpr size_t SZ_WENC  = size_t(2*Lk)*Hk*2;    // bf16 [128][256]
constexpr size_t OFF_WPR  = OFF_WENC + SZ_WENC;
constexpr size_t SZ_WPR   = size_t(2*Lk)*Lk*2;    // bf16 [128][64]
constexpr size_t OFF_EMB  = OFF_WPR + SZ_WPR;     // f32 [B][T][E]

// ---- dynamic LDS layout (bytes), ~299KB of the 320KB/WGP ----
constexpr int SH_HA   = 0;                         // bf16 h buf A  128x256
constexpr int SH_HB   = SH_HA   + Pk*Hk*2;         // bf16 h buf B
constexpr int SH_INP  = SH_HB   + Pk*Hk*2;         // bf16 inp_cat 128x160
constexpr int SH_MUPR = SH_INP  + Pk*KIN*2;        // f32 128x64
constexpr int SH_SGPR = SH_MUPR + Pk*Lk*4;         // f32 128x64
constexpr int SH_Z    = SH_SGPR + Pk*Lk*4;         // f32 128x64
constexpr int SH_ZT   = SH_Z    + Pk*Lk*4;         // f32 128x64 (resample tmp)
constexpr int SH_LA   = SH_ZT   + Pk*Lk*4;         // f32 [128] log_alpha
constexpr int SH_LW   = SH_LA   + Pk*4;            // f32 [128] log_w
constexpr int SH_LPP  = SH_LW   + Pk*4;            // f32 [128]
constexpr int SH_RED  = SH_LPP  + Pk*4;            // f32 [128] reduce scratch
constexpr int SH_CDF  = SH_RED  + Pk*4;            // f32 [128]
constexpr int SH_IDX  = SH_CDF  + Pk*4;            // i32 [128]
constexpr int SH_SCAL = SH_IDX  + Pk*4;            // f32 [16]: lse, logp, need
constexpr int SMEM_BYTES = SH_SCAL + 64;
} // namespace

// ------------------------- helpers -------------------------
DEVI unsigned short f2bf(float f){
  unsigned u = __float_as_uint(f);
  u += 0x7fffu + ((u >> 16) & 1u);
  return (unsigned short)(u >> 16);
}
DEVI float bf2f(unsigned short u){ return __uint_as_float(((unsigned)u) << 16); }

DEVI v8f vzero(){ v8f z = {0.f,0.f,0.f,0.f,0.f,0.f,0.f,0.f}; return z; }

union TileU { uint4 q[2]; v16bf v; };

// A-tile (16x32 bf16) from LDS, row-major, per ISA 16-bit A layout:
// lanes 0-15 row M, K {0..7,16..23}; lanes 16-31 K {8..15,24..31}
DEVI v16bf load_a_lds(const unsigned short* base, int row0, int k0, int stride){
  const int lane = threadIdx.x & 31;
  const unsigned short* p = base + (size_t)(row0 + (lane & 15))*stride + k0 + (lane >> 4)*8;
  TileU t;
  t.q[0] = *(const uint4*)(p);
  t.q[1] = *(const uint4*)(p + 16);
  return t.v;
}
// B-tile (32x16 bf16) from global weights stored [N][K]:
// lanes 0-15 col N, K {0..7,16..23}; lanes 16-31 K {8..15,24..31}
DEVI v16bf load_b_glb(const unsigned short* base, int k0, int n0, int strideK){
  const int lane = threadIdx.x & 31;
  const unsigned short* p = base + (size_t)(n0 + (lane & 15))*strideK + k0 + (lane >> 4)*8;
  TileU t;
  t.q[0] = *(const uint4*)(p);
  t.q[1] = *(const uint4*)(p + 16);
  return t.v;
}
DEVI v8f wmma_bf16(v16bf a, v16bf b, v8f c){
  return __builtin_amdgcn_wmma_f32_16x16x32_bf16(false, a, false, b, (short)0, c, false, false);
}

DEVI float sigmoidf_(float x){ return 1.f/(1.f + __expf(-x)); }
// branch-free stable softplus: max(x,0) + log(1 + exp(-|x|))
DEVI float softplusf_(float x){
  return fmaxf(x, 0.f) + __logf(1.f + __expf(-fabsf(x)));
}

DEVI unsigned hash_u32(unsigned a){
  a ^= a >> 16; a *= 0x7feb352du; a ^= a >> 15; a *= 0x846ca68bu; a ^= a >> 16; return a;
}
DEVI unsigned mixseed(int t, int b, int r, int c, unsigned salt){
  return (unsigned)t*0x9E3779B1u ^ (unsigned)b*0x85EBCA77u ^
         (unsigned)r*0xC2B2AE3Du ^ (unsigned)c*0x27D4EB2Fu ^ salt;
}
DEVI float hash_unif(unsigned s){ return (hash_u32(s) >> 8) * 5.9604645e-08f; }
DEVI float hash_normal(unsigned s){
  unsigned h1 = hash_u32(s), h2 = hash_u32(s ^ 0x6a09e667u);
  float u1 = (h1 >> 8) * 5.9604645e-08f + 1e-9f;
  float u2 = (h2 >> 8) * 5.9604645e-08f;
  return sqrtf(-2.f*__logf(u1)) * __cosf(6.28318530718f*u2);
}

// ------------------------- prep kernels -------------------------
__global__ void kConvPad(const float* __restrict__ src, unsigned short* __restrict__ dst,
                         int N, int Ksrc, int Kdst){
  int i = blockIdx.x*blockDim.x + threadIdx.x;
  if (i >= N*Kdst) return;
  int n = i / Kdst, k = i - n*Kdst;
  dst[i] = f2bf((k < Ksrc) ? src[(size_t)n*Ksrc + k] : 0.f);
}
__global__ void kConvT(const float* __restrict__ src, unsigned short* __restrict__ dst,
                       int K, int N){  // src [K][N] -> dst [N][K]
  int i = blockIdx.x*blockDim.x + threadIdx.x;
  if (i >= K*N) return;
  int n = i / K, k = i - n*K;
  dst[i] = f2bf(src[(size_t)k*N + n]);
}
__global__ void kEmbed(const float* __restrict__ x, const float* __restrict__ W,
                       const float* __restrict__ b, float* __restrict__ e){
  int i = blockIdx.x*blockDim.x + threadIdx.x;
  if (i >= Bk*Tk*Ek) return;
  int bt = i >> 6, c = i & 63;
  e[i] = b[c] + x[bt*2+0]*W[0*Ek + c] + x[bt*2+1]*W[1*Ek + c];
}

// ------------------------- main persistent FIVO kernel -------------------------
__global__ __launch_bounds__(256, 1)
void fivo_main(const float* __restrict__ y,         // [B,T,D]
               const float* __restrict__ emb,       // [B,T,E]
               const unsigned short* __restrict__ wih,   // bf16 [768][160]
               const unsigned short* __restrict__ whh,   // bf16 [768][256]
               const unsigned short* __restrict__ wenc,  // bf16 [128][256]
               const unsigned short* __restrict__ wpr,   // bf16 [128][64]
               const float* __restrict__ wdec,      // f32 [64][4]
               const float* __restrict__ bih, const float* __restrict__ bhh,
               const float* __restrict__ benc, const float* __restrict__ bpr,
               const float* __restrict__ bdec,
               float* __restrict__ out)
{
  extern __shared__ char smem[];
  unsigned short* sHA  = (unsigned short*)(smem + SH_HA);
  unsigned short* sHB  = (unsigned short*)(smem + SH_HB);
  unsigned short* sInp = (unsigned short*)(smem + SH_INP);
  float* sMuPr = (float*)(smem + SH_MUPR);
  float* sSgPr = (float*)(smem + SH_SGPR);
  float* sZ    = (float*)(smem + SH_Z);
  float* sZt   = (float*)(smem + SH_ZT);
  float* sLA   = (float*)(smem + SH_LA);
  float* sLW   = (float*)(smem + SH_LW);
  float* sLPP  = (float*)(smem + SH_LPP);
  float* sRed  = (float*)(smem + SH_RED);
  float* sCdf  = (float*)(smem + SH_CDF);
  int*   sIdx  = (int*)  (smem + SH_IDX);
  float* sScal = (float*)(smem + SH_SCAL);

  const int tid  = threadIdx.x;
  const int bid  = blockIdx.x;
  const int wv   = tid >> 5;        // wave 0..7 (wave32): owns particle rows 16wv..16wv+15
  const int lane = tid & 31;
  const int half = lane >> 4;
  const int nIdx = lane & 15;

  // ---- init persistent state ----
  for (int e = tid; e < Pk*Hk; e += 256) sHA[e] = 0;       // h = 0 (bf16 0)
  for (int e = tid; e < Pk*Lk; e += 256) sZ[e] = 0.f;      // z = 0
  if (tid < Pk) sLW[tid] = 0.f;                            // log_w = 0
  if (tid == 0){ sScal[0] = 0.f; sScal[1] = 0.f; sScal[2] = 0.f; }
  for (int e = tid; e < Pk*(KIN - INk); e += 256){          // zero K padding once
    int r = e / (KIN - INk), c = e - r*(KIN - INk);
    sInp[r*KIN + INk + c] = 0;
  }
  __syncthreads();

  #pragma unroll 1
  for (int t = 0; t < Tk; ++t){
    unsigned short* hCur = (t & 1) ? sHB : sHA;
    unsigned short* hNew = (t & 1) ? sHA : sHB;

    if (lane == 0){  // pull this step's weight stream into cache (global_prefetch_b8)
      __builtin_prefetch(whh + (size_t)wv * (3*Hk*Hk/8), 0, 3);
      __builtin_prefetch(wih + (size_t)wv * (3*Hk*KIN/8), 0, 3);
    }

    // ---- build inp_cat = [xe | z | y] as bf16, zero log_alpha ----
    {
      const float* et = emb + ((size_t)bid*Tk + t)*Ek;
      for (int e = tid; e < Pk*Ek; e += 256){
        int r = e >> 6, c = e & 63;
        sInp[r*KIN + c] = f2bf(et[c]);
      }
      for (int e = tid; e < Pk*Lk; e += 256){
        int r = e >> 6, c = e & 63;
        sInp[r*KIN + Ek + c] = f2bf(sZ[r*Lk + c]);
      }
      const float* yt = y + ((size_t)bid*Tk + t)*Dk;
      for (int e = tid; e < Pk*Dk; e += 256){
        int r = e >> 1, c = e & 1;
        sInp[r*KIN + Ek + Lk + c] = f2bf(yt[c]);
      }
      if (tid < Pk) sLA[tid] = 0.f;
    }
    __syncthreads();

    // ---- prior: z @ W_pr  ([128,64]@[64,128]), mu tile jj / sig tile jj+4
    // paired so the epilogue is branch-free (uniform per tile) ----
    #pragma unroll 1
    for (int jj = 0; jj < 4; ++jj){
      v8f accMu = vzero(), accSr = vzero();
      for (int kc = 0; kc < 2; ++kc){
        v16bf a = load_a_lds(sInp, wv*16, Ek + kc*32, KIN);
        accMu = wmma_bf16(a, load_b_glb(wpr, kc*32,  jj*16,    Lk), accMu);
        accSr = wmma_bf16(a, load_b_glb(wpr, kc*32, (jj+4)*16, Lk), accSr);
      }
      const int col = jj*16 + nIdx;            // 0..63
      const float bMu = bpr[col], bSr = bpr[Lk + col];
      for (int k = 0; k < 8; ++k){
        int row = wv*16 + k + 8*half;
        sMuPr[row*Lk + col] = accMu[k] + bMu;
        sSgPr[row*Lk + col] = softplusf_(accSr[k] + bSr + SPB);
      }
    }

    // ---- GRU: gi = inp@W_ih.T, gh = h@W_hh.T, fused gate math ----
    // A-tiles streamed from LDS inside the K loops: live state is
    // 4 accumulators + in-flight tiles (no scratch spills).
    #pragma unroll 1
    for (int j = 0; j < 16; ++j){
      v8f aR = vzero(), aU = vzero(), aIN = vzero(), aHN = vzero();
      for (int kc = 0; kc < 5; ++kc){
        v16bf a = load_a_lds(sInp, wv*16, kc*32, KIN);
        aR  = wmma_bf16(a, load_b_glb(wih, kc*32, 0*Hk + j*16, KIN), aR);
        aU  = wmma_bf16(a, load_b_glb(wih, kc*32, 1*Hk + j*16, KIN), aU);
        aIN = wmma_bf16(a, load_b_glb(wih, kc*32, 2*Hk + j*16, KIN), aIN);
      }
      for (int kc = 0; kc < 8; ++kc){
        v16bf a = load_a_lds(hCur, wv*16, kc*32, Hk);
        aR  = wmma_bf16(a, load_b_glb(whh, kc*32, 0*Hk + j*16, Hk), aR);
        aU  = wmma_bf16(a, load_b_glb(whh, kc*32, 1*Hk + j*16, Hk), aU);
        aHN = wmma_bf16(a, load_b_glb(whh, kc*32, 2*Hk + j*16, Hk), aHN);
      }
      const int col = j*16 + nIdx;
      const float bR  = bih[col]      + bhh[col];
      const float bU  = bih[Hk+col]   + bhh[Hk+col];
      const float bIN = bih[2*Hk+col];
      const float bHN = bhh[2*Hk+col];
      for (int k = 0; k < 8; ++k){
        int row = wv*16 + k + 8*half;
        float r  = sigmoidf_(aR[k] + bR);
        float u  = sigmoidf_(aU[k] + bU);
        float nn = tanhf(aIN[k] + bIN + r*(aHN[k] + bHN));
        float hold = bf2f(hCur[row*Hk + col]);
        hNew[row*Hk + col] = f2bf((1.f - u)*nn + u*hold);
      }
    }
    __syncthreads();

    // ---- encoder: h_new @ W_enc; mu tile (jj) and sr tile (jj+4) paired so
    // the reparameterized sample needs only 2 live accumulators ----
    #pragma unroll 1
    for (int jj = 0; jj < 4; ++jj){
      v8f accMu = vzero(), accSr = vzero();
      for (int kc = 0; kc < 8; ++kc){
        v16bf a = load_a_lds(hNew, wv*16, kc*32, Hk);
        accMu = wmma_bf16(a, load_b_glb(wenc, kc*32,  jj*16,    Hk), accMu);
        accSr = wmma_bf16(a, load_b_glb(wenc, kc*32, (jj+4)*16, Hk), accSr);
      }
      const int col = jj*16 + nIdx;            // 0..63
      const float bMu = benc[col], bSr = benc[Lk + col];
      for (int k = 0; k < 8; ++k){
        int row = wv*16 + k + 8*half;
        float mu = accMu[k] + bMu;
        float sr = accSr[k] + bSr;
        float sg = softplusf_(sr + SPB);
        float eps = hash_normal(mixseed(t, bid, row, col, 0xA5A5A5A5u));
        float z = mu + sg*eps;
        sZ[row*Lk + col] = z;
        float lpPo = -0.5f*eps*eps - __logf(sg);          // const cancels with prior
        float dz = (z - sMuPr[row*Lk + col]) / sSgPr[row*Lk + col];
        float lpPr = -0.5f*dz*dz - __logf(sSgPr[row*Lk + col]);
        atomicAdd(&sLA[row], lpPr - lpPo);                 // ds_add_f32
      }
    }
    __syncthreads();

    // ---- decoder (tiny GEMV per particle) + y logprob + lpp ----
    if (tid < Pk){
      float m0 = bdec[0], m1 = bdec[1], s0 = bdec[2], s1 = bdec[3];
      for (int l = 0; l < Lk; ++l){
        float zv = sZ[tid*Lk + l];
        m0 += zv * wdec[l*4 + 0];  m1 += zv * wdec[l*4 + 1];
        s0 += zv * wdec[l*4 + 2];  s1 += zv * wdec[l*4 + 3];
      }
      float sg0 = softplusf_(s0 + SPB), sg1 = softplusf_(s1 + SPB);
      const float* yt = y + ((size_t)bid*Tk + t)*Dk;
      float d0 = (yt[0] - m0)/sg0, d1 = (yt[1] - m1)/sg1;
      float lpY = -0.5f*d0*d0 - __logf(sg0) - 0.5f*LOG2PI
                - 0.5f*d1*d1 - __logf(sg1) - 0.5f*LOG2PI;
      float lpp = sLW[tid] + sLA[tid] + lpY;
      sLPP[tid] = lpp;
      sRed[tid] = lpp;
    }
    // ---- logsumexp over 128 particles ----
    for (int off = 64; off > 0; off >>= 1){
      __syncthreads();
      if (tid < off) sRed[tid] = fmaxf(sRed[tid], sRed[tid + off]);
    }
    __syncthreads();
    float mx = sRed[0];
    __syncthreads();
    if (tid < Pk) sRed[tid] = __expf(sLPP[tid] - mx);
    for (int off = 64; off > 0; off >>= 1){
      __syncthreads();
      if (tid < off) sRed[tid] += sRed[tid + off];
    }
    __syncthreads();
    if (tid == 0){
      float lse = mx + __logf(sRed[0]);
      sScal[0] = lse;
      sScal[1] += lse;           // log_p accumulation
    }
    __syncthreads();
    if (tid < Pk){
      float lw = sLPP[tid] - sScal[0];
      sLW[tid] = lw;
      sRed[tid] = __expf(2.f*lw);
    }
    for (int off = 64; off > 0; off >>= 1){
      __syncthreads();
      if (tid < off) sRed[tid] += sRed[tid + off];
    }
    __syncthreads();
    if (tid == 0) sScal[2] = (1.f/sRed[0] < 0.5f*(float)Pk) ? 1.f : 0.f; // ESS < P/2
    __syncthreads();

    // ---- multinomial resampling (block-uniform branch) ----
    if (sScal[2] > 0.5f){
      if (tid < Pk) sCdf[tid] = __expf(sLW[tid]);
      for (int off = 1; off < Pk; off <<= 1){     // Hillis-Steele inclusive scan
        __syncthreads();
        float v = 0.f;
        if (tid < Pk){ v = sCdf[tid]; if (tid >= off) v += sCdf[tid - off]; }
        __syncthreads();
        if (tid < Pk) sCdf[tid] = v;
      }
      __syncthreads();
      if (tid < Pk){
        float u = hash_unif(mixseed(t, bid, tid, 777, 0x5EEDFACEu)) * sCdf[Pk - 1];
        int lo = 0, hi = Pk - 1;
        while (lo < hi){ int mid = (lo + hi) >> 1; if (sCdf[mid] < u) lo = mid + 1; else hi = mid; }
        sIdx[tid] = lo;
        sLW[tid] = -LOGP;
      }
      __syncthreads();
      for (int e = tid; e < Pk*Lk; e += 256){
        int r = e >> 6, c = e & 63;
        sZt[e] = sZ[sIdx[r]*Lk + c];
      }
      __syncthreads();
      for (int e = tid; e < Pk*Lk; e += 256) sZ[e] = sZt[e];
    }
    __syncthreads();
  } // t

  if (tid == 0) out[bid] = sScal[1];
}

// ------------------------- launcher -------------------------
extern "C" void kernel_launch(void* const* d_in, const int* in_sizes, int n_in,
                              void* d_out, int out_size, void* d_ws, size_t ws_size,
                              hipStream_t stream){
  (void)in_sizes; (void)n_in; (void)out_size; (void)ws_size;
  const float* x     = (const float*)d_in[0];
  const float* y     = (const float*)d_in[1];
  const float* W_emb = (const float*)d_in[2];
  const float* b_emb = (const float*)d_in[3];
  const float* W_ih  = (const float*)d_in[4];
  const float* W_hh  = (const float*)d_in[5];
  const float* b_ih  = (const float*)d_in[6];
  const float* b_hh  = (const float*)d_in[7];
  const float* W_enc = (const float*)d_in[8];
  const float* b_enc = (const float*)d_in[9];
  const float* W_pr  = (const float*)d_in[10];
  const float* b_pr  = (const float*)d_in[11];
  const float* W_dec = (const float*)d_in[12];
  const float* b_dec = (const float*)d_in[13];
  float* out = (float*)d_out;

  char* ws = (char*)d_ws;
  unsigned short* wih  = (unsigned short*)(ws + OFF_WIH);
  unsigned short* whh  = (unsigned short*)(ws + OFF_WHH);
  unsigned short* wenc = (unsigned short*)(ws + OFF_WENC);
  unsigned short* wpr  = (unsigned short*)(ws + OFF_WPR);
  float*          emb  = (float*)(ws + OFF_EMB);

  { int n = 3*Hk*KIN; kConvPad<<<(n+255)/256, 256, 0, stream>>>(W_ih, wih, 3*Hk, INk, KIN); }
  { int n = 3*Hk*Hk;  kConvPad<<<(n+255)/256, 256, 0, stream>>>(W_hh, whh, 3*Hk, Hk, Hk); }
  { int n = 2*Lk*Hk;  kConvT  <<<(n+255)/256, 256, 0, stream>>>(W_enc, wenc, Hk, 2*Lk); }
  { int n = 2*Lk*Lk;  kConvT  <<<(n+255)/256, 256, 0, stream>>>(W_pr,  wpr,  Lk, 2*Lk); }
  { int n = Bk*Tk*Ek; kEmbed  <<<(n+255)/256, 256, 0, stream>>>(x, W_emb, b_emb, emb); }

  hipFuncSetAttribute(reinterpret_cast<const void*>(fivo_main),
                      hipFuncAttributeMaxDynamicSharedMemorySize, SMEM_BYTES);
  fivo_main<<<dim3(Bk), dim3(256), SMEM_BYTES, stream>>>(
      y, emb, wih, whh, wenc, wpr, W_dec, b_ih, b_hh, b_enc, b_pr, b_dec, out);
}
